// MutualInformation_17282948399309
// MI455X (gfx1250) — 8-bit matrix operand order repaired and checked by reading, compiled, not run on hardware
//
#include <hip/hip_runtime.h>
#include <hip/hip_bf16.h>
#include <stdint.h>

#define BATCH 32768
#define NBITS 512

typedef int v8i __attribute__((ext_vector_type(8)));

#define OFF_C    (16u * 1024u * 1024u)
#define OFF_CNT  (OFF_C + 1024u * 1024u)
#define OFF_PSUM (OFF_CNT + 4096u)
#define OFF_PCNT (OFF_PSUM + 4096u)

__global__ void k_binarize_transpose(const float* __restrict__ bits,
                                     uint8_t* __restrict__ At) {
    __shared__ uint8_t tile[32][33];
    const int i0 = blockIdx.x * 32;
    const int k0 = blockIdx.y * 32;
    const int tx = threadIdx.x, ty = threadIdx.y;
#pragma unroll
    for (int r = 0; r < 4; ++r) {
        const int kk = ty + r * 8;
        const float v = bits[(size_t)(k0 + kk) * NBITS + i0 + tx];
        tile[kk][tx] = (v > 0.0f) ? (uint8_t)1 : (uint8_t)0;
    }
    __syncthreads();
#pragma unroll
    for (int r = 0; r < 4; ++r) {
        const int ii = ty + r * 8;
        At[(size_t)(i0 + ii) * BATCH + k0 + tx] = tile[tx][ii];
    }
}

__global__ void k_colcount(const uint8_t* __restrict__ At, int* __restrict__ cnt) {
    __shared__ int red[256];
    const int i = blockIdx.x;
    const unsigned long long* p =
        (const unsigned long long*)(At + (size_t)i * BATCH);
    int s = 0;
    for (int idx = threadIdx.x; idx < BATCH / 8; idx += 256)
        s += __popcll(p[idx]);
    red[threadIdx.x] = s;
    __syncthreads();
    for (int w = 128; w > 0; w >>= 1) {
        if (threadIdx.x < w) red[threadIdx.x] += red[threadIdx.x + w];
        __syncthreads();
    }
    if (threadIdx.x == 0) cnt[i] = red[0];
}

__global__ void k_gram_wmma(const uint8_t* __restrict__ At, int* __restrict__ C) {
    const int ib = blockIdx.x, jb = blockIdx.y;
    if (jb > ib) return;

    __shared__ int acc_s[256];
    const int tid  = threadIdx.x;
    acc_s[tid] = 0;
    __syncthreads();

    const int wave   = tid >> 5;
    const int lane   = tid & 31;
    const int lanelo = lane & 15;
    const int hi     = lane >> 4;
    const int i0 = ib * 16, j0 = jb * 16;

    const uint8_t* __restrict__ pa =
        At + (size_t)(i0 + lanelo) * BATCH + hi * 8;
    const uint8_t* __restrict__ pb =
        At + (size_t)(j0 + lanelo) * BATCH + hi * 8;

    v8i acc = {};
    const int kbase = wave * (BATCH / 8);
#pragma unroll 4
    for (int t = 0; t < (BATCH / 8) / 64; ++t) {
        const int k0 = kbase + t * 64;
        v8i a, b;
        const int2 a0 = *(const int2*)(pa + k0 +  0);
        const int2 a1 = *(const int2*)(pa + k0 + 16);
        const int2 a2 = *(const int2*)(pa + k0 + 32);
        const int2 a3 = *(const int2*)(pa + k0 + 48);
        a[0]=a0.x; a[1]=a0.y; a[2]=a1.x; a[3]=a1.y;
        a[4]=a2.x; a[5]=a2.y; a[6]=a3.x; a[7]=a3.y;
        const int2 b0 = *(const int2*)(pb + k0 +  0);
        const int2 b1 = *(const int2*)(pb + k0 + 16);
        const int2 b2 = *(const int2*)(pb + k0 + 32);
        const int2 b3 = *(const int2*)(pb + k0 + 48);
        b[0]=b0.x; b[1]=b0.y; b[2]=b1.x; b[3]=b1.y;
        b[4]=b2.x; b[5]=b2.y; b[6]=b3.x; b[7]=b3.y;
        acc = __builtin_amdgcn_wmma_i32_16x16x64_iu8(
             false, a,  false, b, acc,
             false,  false);
    }

#pragma unroll
    for (int r = 0; r < 8; ++r) {
        const int M = r + 8 * hi;
        atomicAdd(&acc_s[M * 16 + lanelo], acc[r]);
    }
    __syncthreads();

    const int M = tid >> 4, N = tid & 15;
    C[(size_t)(i0 + M) * NBITS + (j0 + N)] = acc_s[tid];
}

__global__ void k_mi_terms(const int* __restrict__ C, const int* __restrict__ cnt,
                           float* __restrict__ psum, int* __restrict__ pcnt) {
    __shared__ float sred[256];
    __shared__ int   cred[256];
    const int g = blockIdx.x * 256 + threadIdx.x;
    const int i = g >> 9, j = g & 511;
    float s = 0.0f; int c = 0;
    if (i > j) {
        const int n11 = C[(size_t)i * NBITS + j];
        const int ci = cnt[i], cj = cnt[j];
        int nn[4];
        nn[3] = n11;
        nn[2] = ci - n11;
        nn[1] = cj - n11;
        nn[0] = BATCH - ci - cj + n11;
        const int mi[2] = {BATCH - ci, ci};
        const int mj[2] = {BATCH - cj, cj};
        const float invB = 1.0f / (float)BATCH;
#pragma unroll
        for (int m = 0; m < 2; ++m)
#pragma unroll
            for (int n = 0; n < 2; ++n) {
                const int nv = nn[m * 2 + n];
                if (nv > 0 && mi[m] > 0 && mj[n] > 0) {
                    const float p = (float)nv * invB;
                    const float ratio = ((float)nv * (float)BATCH) /
                                        ((float)mi[m] * (float)mj[n]);
                    s += p * logf(ratio);
                    c += 1;
                }
            }
    }
    sred[threadIdx.x] = s; cred[threadIdx.x] = c;
    __syncthreads();
    for (int w = 128; w > 0; w >>= 1) {
        if (threadIdx.x < w) {
            sred[threadIdx.x] += sred[threadIdx.x + w];
            cred[threadIdx.x] += cred[threadIdx.x + w];
        }
        __syncthreads();
    }
    if (threadIdx.x == 0) { psum[blockIdx.x] = sred[0]; pcnt[blockIdx.x] = cred[0]; }
}

__global__ void k_finalize(const float* __restrict__ psum,
                           const int* __restrict__ pcnt,
                           float* __restrict__ out) {
    __shared__ float sred[256];
    __shared__ int   cred[256];
    float s = 0.0f; int c = 0;
    for (int idx = threadIdx.x; idx < 1024; idx += 256) {
        s += psum[idx]; c += pcnt[idx];
    }
    sred[threadIdx.x] = s; cred[threadIdx.x] = c;
    __syncthreads();
    for (int w = 128; w > 0; w >>= 1) {
        if (threadIdx.x < w) {
            sred[threadIdx.x] += sred[threadIdx.x + w];
            cred[threadIdx.x] += cred[threadIdx.x + w];
        }
        __syncthreads();
    }
    if (threadIdx.x == 0) out[0] = sred[0] / (float)cred[0];
}

extern "C" void kernel_launch(void* const* d_in, const int* in_sizes, int n_in,
                              void* d_out, int out_size, void* d_ws, size_t ws_size,
                              hipStream_t stream) {
    const float* bits = (const float*)d_in[0];
    uint8_t* At  = (uint8_t*)d_ws;
    int*     C   = (int*)((char*)d_ws + OFF_C);
    int*     cnt = (int*)((char*)d_ws + OFF_CNT);
    float*  psum = (float*)((char*)d_ws + OFF_PSUM);
    int*    pcnt = (int*)((char*)d_ws + OFF_PCNT);
    float*   out = (float*)d_out;

    k_binarize_transpose<<<dim3(NBITS / 32, BATCH / 32), dim3(32, 8), 0, stream>>>(bits, At);
    k_colcount<<<NBITS, 256, 0, stream>>>(At, cnt);
    k_gram_wmma<<<dim3(NBITS / 16, NBITS / 16), 256, 0, stream>>>(At, C);
    k_mi_terms<<<(NBITS * NBITS) / 256, 256, 0, stream>>>(C, cnt, psum, pcnt);
    k_finalize<<<1, 256, 0, stream>>>(psum, pcnt, out);
}
